// GCN2Model_48034914238531
// MI455X (gfx1250) — matrix operand order repaired and checked by reading, not compiled, not run on hardware
//
#include <hip/hip_runtime.h>

// ---------------------------------------------------------------------------
// GCNII (2-layer) inference for MI455X / gfx1250, wave32 + WMMA bf16.
// ---------------------------------------------------------------------------

typedef __attribute__((ext_vector_type(16))) __bf16 bf16x16;
typedef __attribute__((ext_vector_type(8)))  float  f32x8;

#define DFEAT   128
#define NGRAPH  64
#define PREDH   32

__device__ __forceinline__ unsigned short f2bf(float f) {
    unsigned int u = __float_as_uint(f);
    u += 0x7FFFu + ((u >> 16) & 1u);          // round-to-nearest-even
    return (unsigned short)(u >> 16);
}

// ---- degree / norm --------------------------------------------------------
__global__ void gcn2_init_deg(float* deg, int n) {
    int i = blockIdx.x * 256 + threadIdx.x;
    if (i < n) deg[i] = 1.0f;                 // self-loop contributes 1
}

__global__ void gcn2_accum_deg(const int* __restrict__ dst, float* deg, int e) {
    int i = blockIdx.x * 256 + threadIdx.x;
    if (i < e) unsafeAtomicAdd(&deg[dst[i]], 1.0f);
}

__global__ void gcn2_make_norm(const float* __restrict__ deg, float* norm, int n) {
    int i = blockIdx.x * 256 + threadIdx.x;
    if (i < n) norm[i] = rsqrtf(fmaxf(deg[i], 1.0f));
}

// ---- element-wise preps ---------------------------------------------------
// y[i,c] = x[i,c] * norm[i]
__global__ void gcn2_scale_rows(const float* __restrict__ x, const float* __restrict__ norm,
                                float* __restrict__ y, int total) {
    int i = blockIdx.x * 256 + threadIdx.x;
    if (i < total) y[i] = x[i] * norm[i >> 7];
}

// agg := (1-ALPHA) * agg * norm  (hs, f32, in place) ; Ub := bf16(hs)
__global__ void gcn2_prep_hs(float* __restrict__ agg, const float* __restrict__ norm,
                             unsigned short* __restrict__ Ub, int total) {
    int i = blockIdx.x * 256 + threadIdx.x;
    if (i < total) {
        float u = 0.5f * agg[i] * norm[i >> 7];
        agg[i] = u;
        Ub[i] = f2bf(u);
    }
}

// Vb := bf16(ALPHA * feature)   (shared by both layers)
__global__ void gcn2_prep_h0s(const float* __restrict__ feat, unsigned short* __restrict__ Vb,
                              int total) {
    int i = blockIdx.x * 256 + threadIdx.x;
    if (i < total) Vb[i] = f2bf(0.5f * feat[i]);
}

// ---- SpMM scatter-add (edge-parallel, float4 lanes, HW f32 atomics) -------
__global__ void gcn2_spmm_edges(const float* __restrict__ scaled, const int* __restrict__ src,
                                const int* __restrict__ dst, float* __restrict__ agg,
                                long total) {
    long idx = (long)blockIdx.x * 256 + threadIdx.x;
    if (idx >= total) return;
    int e = (int)(idx >> 5);
    int c = ((int)idx & 31) * 4;
    int s = src[e], d = dst[e];
    const float4 v = *(const float4*)(scaled + (size_t)s * DFEAT + c);
    float* p = agg + (size_t)d * DFEAT + c;
    unsafeAtomicAdd(p + 0, v.x);
    unsafeAtomicAdd(p + 1, v.y);
    unsafeAtomicAdd(p + 2, v.z);
    unsafeAtomicAdd(p + 3, v.w);
}

// ---- pack 128x128 f32 weights into per-lane WMMA B-fragment layout --------
// layout index: ct*2048 + kb*512 + lane*16 + h ; element = W[K*128 + n]
// with n = ct*16 + (lane&15), K = kb*32 + 8*(lane>>4) + (h&7) + 16*(h>>3)   (ISA 7.12.2 B)
__global__ void gcn2_pack_weights(const float* __restrict__ w0, const float* __restrict__ w1,
                                  const float* __restrict__ w2, const float* __restrict__ w3,
                                  unsigned short* __restrict__ Wp) {
    int tid = blockIdx.x * 256 + threadIdx.x;
    if (tid >= 4 * 16384) return;
    int m = tid >> 14;
    int r = tid & 16383;
    int h = r & 15;
    int lane = (r >> 4) & 31;
    int kb = (r >> 9) & 3;
    int ct = r >> 11;
    int n = ct * 16 + (lane & 15);
    int K = kb * 32 + (lane >> 4) * 8 + (h & 7) + (h >> 3) * 16;
    const float* w = (m == 0) ? w0 : (m == 1) ? w1 : (m == 2) ? w2 : w3;
    Wp[tid] = f2bf(w[K * DFEAT + n]);
}

// ---- fused GCNII layer GEMM: out = relu((1-b)(U+V) + b(U@W1 + V@W2) + bias)
// one 16x16 tile per wave, 8 waves/block cover all 128 columns of 16 rows.
__global__ __launch_bounds__(256) void gcn2_wmma_gemm(
    const unsigned short* __restrict__ Ub, const unsigned short* __restrict__ Vb,
    const float* __restrict__ Uf,     // hs in f32 (agg, already scaled)
    const float* __restrict__ feat0,  // original features (V = 0.5*feat0)
    const unsigned short* __restrict__ Wp1, const unsigned short* __restrict__ Wp2,
    const float* __restrict__ bias, float beta, float* __restrict__ out, int nrows) {
    __shared__ unsigned short ldsU[16 * DFEAT];
    __shared__ unsigned short ldsV[16 * DFEAT];
    const int tid = threadIdx.x;
    const int rowBase = blockIdx.x * 16;

    {   // cooperative stage of the 16x128 bf16 A-tiles (1024 dwords each)
        const unsigned int* gu = (const unsigned int*)(Ub + (size_t)rowBase * DFEAT);
        const unsigned int* gv = (const unsigned int*)(Vb + (size_t)rowBase * DFEAT);
        unsigned int* lu = (unsigned int*)ldsU;
        unsigned int* lv = (unsigned int*)ldsV;
        for (int k = tid; k < 1024; k += 256) { lu[k] = gu[k]; lv[k] = gv[k]; }
    }
    __syncthreads();

    const int wave = tid >> 5;
    const int lane = tid & 31;
    const int li = lane & 15;
    const int g  = lane >> 4;
    const int ct = wave;                      // column tile 0..7

    union AFrag { uint4 q[2]; bf16x16 v; };

    f32x8 accU = {};
    f32x8 accV = {};
#pragma unroll
    for (int kb = 0; kb < 4; ++kb) {
        AFrag au, av, b1, b2;
        // A fragment (ISA 7.12.2, 16-bit A 16x32): lane(li,g) needs
        // K = kb*32 + 8g + [0..7]  and  K = kb*32 + 8g + 16 + [0..7]
        const uint4* pu = (const uint4*)(ldsU + li * DFEAT + kb * 32 + g * 8);
        const uint4* pv = (const uint4*)(ldsV + li * DFEAT + kb * 32 + g * 8);
        au.q[0] = pu[0]; au.q[1] = pu[2];
        av.q[0] = pv[0]; av.q[1] = pv[2];
        // B fragment: pre-packed contiguous 32 bytes per lane
        const uint4* p1 = (const uint4*)(Wp1 + (size_t)((ct * 4 + kb) * 32 + lane) * 16);
        const uint4* p2 = (const uint4*)(Wp2 + (size_t)((ct * 4 + kb) * 32 + lane) * 16);
        b1.q[0] = p1[0]; b1.q[1] = p1[1];
        b2.q[0] = p2[0]; b2.q[1] = p2[1];

        accU = __builtin_amdgcn_wmma_f32_16x16x32_bf16(false, au.v, false, b1.v,
                                                       (short)0, accU, false, false);
        accV = __builtin_amdgcn_wmma_f32_16x16x32_bf16(false, av.v, false, b2.v,
                                                       (short)0, accV, false, false);
    }

    const int col = ct * 16 + li;
    const float bcol = bias[col];
    const float omb = 1.0f - beta;
#pragma unroll
    for (int r = 0; r < 8; ++r) {
        int row = rowBase + r + 8 * g;        // C/D layout: M = r + 8*laneGroup
        if (row < nrows) {
            size_t idx = (size_t)row * DFEAT + col;
            float u = Uf[idx];
            float v = 0.5f * feat0[idx];      // ALPHA * feat0
            float val = omb * (u + v) + beta * (accU[r] + accV[r]) + bcol;
            out[idx] = fmaxf(val, 0.0f);
        }
    }
}

// ---- pooling --------------------------------------------------------------
__global__ void gcn2_count(const int* __restrict__ gid, float* cnt, int n) {
    int i = blockIdx.x * 256 + threadIdx.x;
    if (i < n) unsafeAtomicAdd(&cnt[gid[i]], 1.0f);
}

__global__ void gcn2_pool_sum(const float* __restrict__ x2, const int* __restrict__ gid,
                              float* __restrict__ pooled, long total) {
    long idx = (long)blockIdx.x * 256 + threadIdx.x;
    if (idx >= total) return;
    int i = (int)(idx >> 5);
    int c = ((int)idx & 31) * 4;
    int gidx = gid[i];
    const float4 v = *(const float4*)(x2 + (size_t)i * DFEAT + c);
    float* p = pooled + (size_t)gidx * DFEAT + c;
    unsafeAtomicAdd(p + 0, v.x);
    unsafeAtomicAdd(p + 1, v.y);
    unsafeAtomicAdd(p + 2, v.z);
    unsafeAtomicAdd(p + 3, v.w);
}

// ---- MLP head (single block) ----------------------------------------------
__global__ __launch_bounds__(256) void gcn2_mlp_head(
    const float* __restrict__ pooled, const float* __restrict__ cnt,
    const float* __restrict__ d1w, const float* __restrict__ d1b,
    const float* __restrict__ d2w, const float* __restrict__ d2b,
    float* __restrict__ out) {
    __shared__ float sP[NGRAPH * DFEAT];
    __shared__ float sH[NGRAPH * PREDH];
    int tid = threadIdx.x;
    for (int k = tid; k < NGRAPH * DFEAT; k += 256) {
        float c = cnt[k >> 7];
        c = (c < 1.0f) ? 1.0f : c;
        sP[k] = pooled[k] / c;
    }
    __syncthreads();
    for (int p = tid; p < NGRAPH * PREDH; p += 256) {
        int b = p >> 5, j = p & 31;
        float acc = d1b[j];
        const float* row = sP + b * DFEAT;
        for (int k = 0; k < DFEAT; ++k) acc = fmaf(row[k], d1w[k * PREDH + j], acc);
        sH[p] = fmaxf(acc, 0.0f);
    }
    __syncthreads();
    if (tid < NGRAPH) {
        float acc = d2b[0];
        for (int j = 0; j < PREDH; ++j) acc = fmaf(sH[tid * PREDH + j], d2w[j], acc);
        out[tid] = 1.0f / (1.0f + __expf(-acc));
    }
}

// ---------------------------------------------------------------------------
extern "C" void kernel_launch(void* const* d_in, const int* in_sizes, int n_in,
                              void* d_out, int out_size, void* d_ws, size_t ws_size,
                              hipStream_t stream) {
    (void)n_in; (void)out_size; (void)ws_size;

    const float* feature = (const float*)d_in[0];
    const int*   src     = (const int*)d_in[1];
    const int*   dst     = (const int*)d_in[2];
    const int*   gid     = (const int*)d_in[3];
    const float* w1_1 = (const float*)d_in[4];
    const float* w2_1 = (const float*)d_in[5];
    const float* b_1  = (const float*)d_in[6];
    const float* w1_2 = (const float*)d_in[7];
    const float* w2_2 = (const float*)d_in[8];
    const float* b_2  = (const float*)d_in[9];
    const float* d1w  = (const float*)d_in[10];
    const float* d1b  = (const float*)d_in[11];
    const float* d2w  = (const float*)d_in[12];
    const float* d2b  = (const float*)d_in[13];

    const int N = in_sizes[0] / DFEAT;   // 100000
    const int E = in_sizes[1];           // 1600000
    const int ND = N * DFEAT;

    const float BETA1 = 0.69314718055994531f;  // log(2)
    const float BETA2 = 0.40546510810816438f;  // log(1.5)

    // ---- workspace carve-up (256B aligned) ----
    char* w = (char*)d_ws;
    auto take = [&](size_t bytes) -> char* {
        char* p = w;
        w += (bytes + 255) & ~(size_t)255;
        return p;
    };
    float*          deg    = (float*)take((size_t)N * 4);
    float*          norm   = (float*)take((size_t)N * 4);
    float*          scaled = (float*)take((size_t)ND * 4);  // reused as x2
    float*          agg    = (float*)take((size_t)ND * 4);
    float*          x1     = (float*)take((size_t)ND * 4);
    unsigned short* Ub     = (unsigned short*)take((size_t)ND * 2);
    unsigned short* Vb     = (unsigned short*)take((size_t)ND * 2);
    unsigned short* Wp     = (unsigned short*)take((size_t)4 * 16384 * 2);
    float*          pooled = (float*)take((size_t)(NGRAPH * DFEAT + NGRAPH) * 4);
    float*          cnt    = pooled + NGRAPH * DFEAT;       // contiguous with pooled

    auto cdiv = [](long a, long b) { return (int)((a + b - 1) / b); };

    // ---- degree / norm ----
    gcn2_init_deg<<<cdiv(N, 256), 256, 0, stream>>>(deg, N);
    gcn2_accum_deg<<<cdiv(E, 256), 256, 0, stream>>>(dst, deg, E);
    gcn2_make_norm<<<cdiv(N, 256), 256, 0, stream>>>(deg, norm, N);

    // ---- shared preps ----
    gcn2_prep_h0s<<<cdiv(ND, 256), 256, 0, stream>>>(feature, Vb, ND);
    gcn2_pack_weights<<<cdiv(4 * 16384, 256), 256, 0, stream>>>(w1_1, w2_1, w1_2, w2_2, Wp);

    const int gemmBlocks = cdiv(N, 16);

    // ---- layer 1 ----
    gcn2_scale_rows<<<cdiv(ND, 256), 256, 0, stream>>>(feature, norm, scaled, ND);
    hipMemcpyAsync(agg, scaled, (size_t)ND * 4, hipMemcpyDeviceToDevice, stream); // self-loop
    gcn2_spmm_edges<<<cdiv((long)E * 32, 256), 256, 0, stream>>>(scaled, src, dst, agg,
                                                                 (long)E * 32);
    gcn2_prep_hs<<<cdiv(ND, 256), 256, 0, stream>>>(agg, norm, Ub, ND);
    gcn2_wmma_gemm<<<gemmBlocks, 256, 0, stream>>>(Ub, Vb, agg, feature,
                                                   Wp, Wp + 16384, b_1, BETA1, x1, N);

    // ---- layer 2 ----
    gcn2_scale_rows<<<cdiv(ND, 256), 256, 0, stream>>>(x1, norm, scaled, ND);
    hipMemcpyAsync(agg, scaled, (size_t)ND * 4, hipMemcpyDeviceToDevice, stream);
    gcn2_spmm_edges<<<cdiv((long)E * 32, 256), 256, 0, stream>>>(scaled, src, dst, agg,
                                                                 (long)E * 32);
    gcn2_prep_hs<<<cdiv(ND, 256), 256, 0, stream>>>(agg, norm, Ub, ND);
    float* x2 = scaled;  // scaled buffer is dead at this point -> reuse for x2
    gcn2_wmma_gemm<<<gemmBlocks, 256, 0, stream>>>(Ub, Vb, agg, feature,
                                                   Wp + 32768, Wp + 49152, b_2, BETA2, x2, N);

    // ---- pooling + MLP head ----
    hipMemsetAsync(pooled, 0, (size_t)(NGRAPH * DFEAT + NGRAPH) * 4, stream);
    gcn2_count<<<cdiv(N, 256), 256, 0, stream>>>(gid, cnt, N);
    gcn2_pool_sum<<<cdiv((long)N * 32, 256), 256, 0, stream>>>(x2, gid, pooled, (long)N * 32);
    gcn2_mlp_head<<<1, 256, 0, stream>>>(pooled, cnt, d1w, d1b, d2w, d2b, (float*)d_out);
}
